// WindowAttention_7584912245275
// MI455X (gfx1250) — matrix operand order repaired and checked by reading, not compiled, not run on hardware
//
#include <hip/hip_runtime.h>

// ---------------------------------------------------------------------------
// Window attention, fused, for MI455X (gfx1250, wave32, WMMA 16x16x32 f16).
// x:[2,128,270,480] f32 -> pad reflect to 272x480 -> 4080 windows of 64 tok.
// ---------------------------------------------------------------------------

typedef __attribute__((ext_vector_type(8)))  _Float16 v8h;
typedef __attribute__((ext_vector_type(16))) _Float16 v16h;
typedef __attribute__((ext_vector_type(8)))  float    v8f;

#define DEV __device__ __forceinline__

// ---- problem constants ----
constexpr int Cc   = 128;
constexpr int Hh   = 270;
constexpr int Ww   = 480;
constexpr int WS   = 8;
constexpr int NHEAD= 4;
constexpr int HD   = 32;           // head dim
constexpr int PT   = 1;            // top pad (reflect), bottom pad = 1, no W pad
constexpr int NWH  = 34;           // 272/8
constexpr int NWW  = 60;           // 480/8
constexpr int NWIN = 2 * NWH * NWW;

// ---- LDS layout (units: halves) ----
constexpr int LDX = 136;                       // 64-row arrays, padded stride
constexpr int LDV = 72;                        // Vt / P stride (64 + pad)
constexpr int LDO = 132;                       // f32 out staging stride
constexpr int OFF_X = 0;                       // sX (64x136) then sVt (128x72)
constexpr int OFF_Q = 9216;                    // sQ 64x136
constexpr int OFF_K = OFF_Q + 64 * LDX;        // sK 64x136
constexpr int OFF_P = OFF_K + 64 * LDX;        // sP 4 heads x 64x72
constexpr int OFF_O = OFF_P + NHEAD * 64 * LDV;// sO 64x136 (attn out, f16)
constexpr int SMEM_HALVES = OFF_O + 64 * LDX;  // 53760 halves
constexpr int SMEM_BYTES  = SMEM_HALVES * 2;   // 107520 B (dynamic LDS)
constexpr int OFF_OUT_H = OFF_Q;               // f32 [64][132] overlays sQ+sK

DEV v16h cat8(v8h lo, v8h hi) {
  return __builtin_shufflevector(lo, hi, 0,1,2,3,4,5,6,7,8,9,10,11,12,13,14,15);
}

// A-matrix fragment (16x32 f16, ISA interleave): lane holds row m0+(lane&15);
// lanes<16: K = k0+{0..7, 16..23}; lanes>=16: K = k0+{8..15, 24..31}.
DEV v16h frag_a(const _Float16* __restrict__ base, int stride, int m0, int k0, int lane) {
  const _Float16* p = base + (m0 + (lane & 15)) * stride + k0 + ((lane >> 4) << 3);
  return cat8(*(const v8h*)p, *(const v8h*)(p + 16));
}

// B-matrix fragment (32x16 f16) from [N][K] storage: lane holds col n0+(lane&15);
// lanes<16: K = k0+{0..7, 16..23}; lanes>=16: K = k0+{8..15, 24..31}.
DEV v16h frag_b(const _Float16* __restrict__ base, int stride, int n0, int k0, int lane) {
  const _Float16* p = base + (n0 + (lane & 15)) * stride + k0 + ((lane >> 4) << 3);
  return cat8(*(const v8h*)p, *(const v8h*)(p + 16));
}

DEV v8f wmma16(v16h a, v16h b, v8f c) {
  return __builtin_amdgcn_wmma_f32_16x16x32_f16(false, a, false, b, (short)0, c,
                                                false, false);
}

// ---------------------------------------------------------------------------
// Weight prep: f32 -> f16 + transpose to [N][K] so B-fragments are two
// contiguous b128 loads.
// ---------------------------------------------------------------------------
extern "C" __global__ void prep_weights(const float* __restrict__ wqkv,
                                        const float* __restrict__ wproj,
                                        _Float16* __restrict__ wqkvT,
                                        _Float16* __restrict__ wprojT) {
  int g = blockIdx.x * blockDim.x + threadIdx.x;
  if (g < 384 * 128) {
    int n = g >> 7, k = g & 127;
    wqkvT[g] = (_Float16)wqkv[k * 384 + n];
  } else {
    int g2 = g - 384 * 128;
    if (g2 < 128 * 128) {
      int n = g2 >> 7, k = g2 & 127;
      wprojT[g2] = (_Float16)wproj[k * 128 + n];
    }
  }
}

// ---------------------------------------------------------------------------
// Fused window attention: one workgroup (8 wave32) per window.
// ---------------------------------------------------------------------------
extern "C" __global__ __launch_bounds__(256)
void win_attn(const float* __restrict__ x,
              const _Float16* __restrict__ wqkvT,
              const float* __restrict__ bqkv,
              const _Float16* __restrict__ wprojT,
              const float* __restrict__ bproj,
              float* __restrict__ out) {
  extern __shared__ _Float16 smem[];
  _Float16* sX   = smem + OFF_X;           // [64][136] window input (f16)
  _Float16* sQ   = smem + OFF_Q;           // [64][136]
  _Float16* sK   = smem + OFF_K;           // [64][136]
  _Float16* sP   = smem + OFF_P;           // [4][64][72] softmax probs
  _Float16* sO   = smem + OFF_O;           // [64][136] attn out (f16)
  _Float16* sVt  = smem + OFF_X;           // [128][72] V transposed (reuse sX)
  float*    sOut = (float*)(smem + OFF_OUT_H); // [64][132] f32 (reuse sQ/sK)

  const int tid  = threadIdx.x;
  const int lane = tid & 31;
  const int wv   = tid >> 5;
  const int nlo  = lane & 15;
  const int hig  = lane >> 4;

  const int wid = blockIdx.x;
  const int bb  = wid / (NWH * NWW);
  const int wy  = (wid / NWW) % NWH;
  const int wx  = wid % NWW;

  // ---- Phase 0: gather window (reflect pad) -> sX as f16 ----
  const float* xb = x + (size_t)bb * Cc * Hh * Ww;
  for (int i = 0; i < 32; ++i) {
    int idx = tid + 256 * i;                 // 8192 = 64 tok * 128 ch
    int t = idx & 63, c = idx >> 6;
    int ty = t >> 3, tx = t & 7;
    int r = wy * WS + ty - PT;
    r = (r < 0) ? -r : r;
    r = (r >= Hh) ? (2 * Hh - 2 - r) : r;
    int cl = wx * WS + tx;                   // no horizontal pad (480 % 8 == 0)
    sX[t * LDX + c] = (_Float16)xb[((size_t)c * Hh + r) * Ww + cl];
  }
  __syncthreads();

  // ---- Phase 1: QKV GEMM  [64x128] @ [128x384]  (96 tiles, 12 per wave) ----
  v8f acc[12] = {};
  #pragma unroll
  for (int kt = 0; kt < 4; ++kt) {
    const int k0 = kt * 32;
    v16h aF[4];
    #pragma unroll
    for (int mt = 0; mt < 4; ++mt) aF[mt] = frag_a(sX, LDX, mt * 16, k0, lane);
    #pragma unroll
    for (int j = 0; j < 3; ++j) {
      const int n0 = (wv * 3 + j) * 16;
      v16h bF = frag_b(wqkvT, 128, n0, k0, lane);
      #pragma unroll
      for (int mt = 0; mt < 4; ++mt) acc[j * 4 + mt] = wmma16(aF[mt], bF, acc[j * 4 + mt]);
    }
  }
  // Q and K tiles to LDS now; V tiles held in regs until sX is dead.
  #pragma unroll
  for (int j = 0; j < 3; ++j) {
    const int n0 = (wv * 3 + j) * 16;
    if (n0 < 256) {
      const float bias = bqkv[n0 + nlo];
      _Float16* dst = (n0 < 128) ? (sQ + n0 + nlo) : (sK + (n0 - 128) + nlo);
      #pragma unroll
      for (int mt = 0; mt < 4; ++mt)
        #pragma unroll
        for (int v = 0; v < 8; ++v)
          dst[(mt * 16 + v + hig * 8) * LDX] = (_Float16)(acc[j * 4 + mt][v] + bias);
    }
  }
  __syncthreads();                       // all sX reads complete
  #pragma unroll
  for (int j = 0; j < 3; ++j) {
    const int n0 = (wv * 3 + j) * 16;
    if (n0 >= 256) {                     // V, stored transposed [ch][token]
      const float bias = bqkv[n0 + nlo];
      const int c = n0 - 256 + nlo;
      #pragma unroll
      for (int mt = 0; mt < 4; ++mt)
        #pragma unroll
        for (int v = 0; v < 8; ++v)
          sVt[c * LDV + mt * 16 + v + hig * 8] = (_Float16)(acc[j * 4 + mt][v] + bias);
    }
  }
  __syncthreads();

  // ---- Phase 2: attention. 2 waves per head; wave owns 32 query rows. ----
  const int h    = wv >> 1;
  const int mh   = wv & 1;
  const int hOff = h * HD;
  _Float16* sPh  = sP + h * 64 * LDV;

  v8f sAcc[2][4] = {};
  #pragma unroll
  for (int mi = 0; mi < 2; ++mi) {
    const int m0 = mh * 32 + mi * 16;
    v16h qF = frag_a(sQ, LDX, m0, hOff, lane);   // K = hd = 32, single frag
    #pragma unroll
    for (int ni = 0; ni < 4; ++ni) {
      v16h kF = frag_b(sK, LDX, ni * 16, hOff, lane);
      sAcc[mi][ni] = wmma16(qF, kF, sAcc[mi][ni]);
    }
  }

  const float scale = 0.17677669529663687f;      // 1/sqrt(32)
  #pragma unroll
  for (int mi = 0; mi < 2; ++mi) {
    #pragma unroll
    for (int v = 0; v < 8; ++v) {
      float s[4];
      #pragma unroll
      for (int ni = 0; ni < 4; ++ni) s[ni] = sAcc[mi][ni][v] * scale;
      float mx = fmaxf(fmaxf(s[0], s[1]), fmaxf(s[2], s[3]));
      #pragma unroll
      for (int off = 1; off < 16; off <<= 1) mx = fmaxf(mx, __shfl_xor(mx, off, 32));
      float e[4], sum = 0.f;
      #pragma unroll
      for (int ni = 0; ni < 4; ++ni) { e[ni] = __expf(s[ni] - mx); sum += e[ni]; }
      #pragma unroll
      for (int off = 1; off < 16; off <<= 1) sum += __shfl_xor(sum, off, 32);
      const float inv = 1.0f / sum;
      const int row = mh * 32 + mi * 16 + v + hig * 8;
      #pragma unroll
      for (int ni = 0; ni < 4; ++ni)
        sPh[row * LDV + ni * 16 + nlo] = (_Float16)(e[ni] * inv);
    }
  }

  // P @ V  (wave consumes only rows it wrote -> no extra barrier)
  v8f oAcc[2][2] = {};
  #pragma unroll
  for (int mi = 0; mi < 2; ++mi) {
    const int m0 = mh * 32 + mi * 16;
    #pragma unroll
    for (int kt = 0; kt < 2; ++kt) {
      v16h pF = frag_a(sPh, LDV, m0, kt * 32, lane);
      #pragma unroll
      for (int ni = 0; ni < 2; ++ni) {
        v16h vF = frag_b(sVt, LDV, hOff + ni * 16, kt * 32, lane);
        oAcc[mi][ni] = wmma16(pF, vF, oAcc[mi][ni]);
      }
    }
  }
  #pragma unroll
  for (int mi = 0; mi < 2; ++mi)
    #pragma unroll
    for (int ni = 0; ni < 2; ++ni)
      #pragma unroll
      for (int v = 0; v < 8; ++v)
        sO[(mh * 32 + mi * 16 + v + hig * 8) * LDX + hOff + ni * 16 + nlo] =
            (_Float16)oAcc[mi][ni][v];
  __syncthreads();

  // ---- Phase 3: proj  [64x128] @ [128x128]  (32 tiles, 4 per wave) ----
  v8f pAcc[4] = {};
  #pragma unroll
  for (int kt = 0; kt < 4; ++kt) {
    const int k0 = kt * 32;
    #pragma unroll
    for (int j = 0; j < 4; ++j) {
      const int tt = wv * 4 + j;
      const int mt = tt >> 3, nt = tt & 7;
      v16h aF = frag_a(sO, LDX, mt * 16, k0, lane);
      v16h bF = frag_b(wprojT, 128, nt * 16, k0, lane);
      pAcc[j] = wmma16(aF, bF, pAcc[j]);
    }
  }
  #pragma unroll
  for (int j = 0; j < 4; ++j) {
    const int tt = wv * 4 + j;
    const int mt = tt >> 3, nt = tt & 7;
    const float bias = bproj[nt * 16 + nlo];
    #pragma unroll
    for (int v = 0; v < 8; ++v)
      sOut[(mt * 16 + v + hig * 8) * LDO + nt * 16 + nlo] = pAcc[j][v] + bias;
  }
  __syncthreads();

  // ---- Phase 4: cropped, coalesced NCHW writeout ----
  float* ob = out + (size_t)bb * Cc * Hh * Ww;
  for (int i = 0; i < 32; ++i) {
    int idx = tid + 256 * i;
    int t = idx & 63, c = idx >> 6;
    int ty = t >> 3, tx = t & 7;
    int r  = wy * WS + ty - PT;
    int cl = wx * WS + tx;
    if (r >= 0 && r < Hh)
      ob[((size_t)c * Hh + r) * Ww + cl] = sOut[t * LDO + c];
  }
}

// ---------------------------------------------------------------------------
extern "C" void kernel_launch(void* const* d_in, const int* in_sizes, int n_in,
                              void* d_out, int out_size, void* d_ws, size_t ws_size,
                              hipStream_t stream) {
  const float* x      = (const float*)d_in[0];
  const float* w_qkv  = (const float*)d_in[1];
  const float* b_qkv  = (const float*)d_in[2];
  const float* w_proj = (const float*)d_in[3];
  const float* b_proj = (const float*)d_in[4];
  float* out = (float*)d_out;

  _Float16* wqkvT  = (_Float16*)d_ws;
  _Float16* wprojT = (_Float16*)((char*)d_ws + (size_t)384 * 128 * sizeof(_Float16));

  prep_weights<<<256, 256, 0, stream>>>(w_qkv, w_proj, wqkvT, wprojT);
  win_attn<<<NWIN, 256, SMEM_BYTES, stream>>>(x, wqkvT, b_qkv, wprojT, b_proj, out);
}
